// MagLapNet_35450660061792
// MI455X (gfx1250) — matrix operand order repaired and checked by reading, not compiled, not run on hardware
//
#include <hip/hip_runtime.h>
#include <hip/hip_bf16.h>

typedef __attribute__((ext_vector_type(16))) _Float16 v16h;
typedef __attribute__((ext_vector_type(8)))  _Float16 v8h;
typedef __attribute__((ext_vector_type(8)))  float    v8f;

#define NB     4
#define NNODES 4096
#define KEIG   16
#define NE     32768
#define DM     64
#define DAGGR  256

// CDNA5 16-bit A fragment (ISA 7.12.2): halves h=0..7 -> K = hi*8 + h,
// h=8..15 -> K = 16 + hi*8 + (h-8): two contiguous 8-half (16B) runs.
// B fragment: lane col N = lane&15, element h -> K row = s*32 + K of A's element h:
// two 8-row (16B) runs when weights are stored transposed ([col][krow]).
// C/D: lane col N = lane&15, element i -> row M = i + hi*8.

__device__ __forceinline__ v8f wmma_f16(const v16h& a, const v16h& b, const v8f& c) {
    return __builtin_amdgcn_wmma_f32_16x16x32_f16(false, a, false, b, (short)0, c,
                                                  false, false);
}

__device__ __forceinline__ v16h cat8(v8h lo, v8h hi8) {
    return __builtin_shufflevector(lo, hi8, 0, 1, 2, 3, 4, 5, 6, 7,
                                            8, 9, 10, 11, 12, 13, 14, 15);
}

// four float4 chunks -> one f16 A fragment (v_cvt_pk_f16_f32 pairs)
__device__ __forceinline__ v16h cvt16(float4 a, float4 b, float4 c, float4 d) {
    v16h r;
    r[0]  = (_Float16)a.x; r[1]  = (_Float16)a.y; r[2]  = (_Float16)a.z; r[3]  = (_Float16)a.w;
    r[4]  = (_Float16)b.x; r[5]  = (_Float16)b.y; r[6]  = (_Float16)b.z; r[7]  = (_Float16)b.w;
    r[8]  = (_Float16)c.x; r[9]  = (_Float16)c.y; r[10] = (_Float16)c.z; r[11] = (_Float16)c.w;
    r[12] = (_Float16)d.x; r[13] = (_Float16)d.y; r[14] = (_Float16)d.z; r[15] = (_Float16)d.w;
    return r;
}

// =====================================================================
// Kernel 1: edge MLP + segment-sum scatter (global_atomic_add_f32)
// 256 threads = 4 edges x 64 feature lanes
// =====================================================================
__global__ void __launch_bounds__(256)
edge_kernel(const float* __restrict__ evec,      // [B,N,K]
            const int*   __restrict__ senders,   // [B,E]
            const int*   __restrict__ receivers, // [B,E]
            const float* __restrict__ We,        // [2,64]
            const float* __restrict__ be,        // [64]
            float* __restrict__ sent_agg,        // [B,N,K,64]
            float* __restrict__ recv_agg,        // [B,N,K,64]
            float sign) {
    int idx = blockIdx.x * 4 + (threadIdx.x >> 6);   // edge slot in [0, B*E)
    int d   = threadIdx.x & 63;
    int b   = idx >> 15;                             // E = 32768
    int e   = idx & (NE - 1);

    int s = senders[b * NE + e];
    int r = receivers[b * NE + e];

    float w0 = We[d], w1 = We[DM + d], bb = be[d];
    const float* ev = evec + (size_t)b * NNODES * KEIG;

    // hoist the 16 eigvec values per endpoint: 4x global_load_b128 each
    float sva[KEIG], rva[KEIG];
    const float4* sv4 = (const float4*)(ev + (size_t)s * KEIG);
    const float4* rv4 = (const float4*)(ev + (size_t)r * KEIG);
#pragma unroll
    for (int q = 0; q < 4; ++q) {
        float4 t1 = sv4[q], t2 = rv4[q];
        sva[4 * q + 0] = t1.x; sva[4 * q + 1] = t1.y;
        sva[4 * q + 2] = t1.z; sva[4 * q + 3] = t1.w;
        rva[4 * q + 0] = t2.x; rva[4 * q + 1] = t2.y;
        rva[4 * q + 2] = t2.z; rva[4 * q + 3] = t2.w;
    }
    sva[0] = fabsf(sva[0]); rva[0] = fabsf(rva[0]);
#pragma unroll
    for (int k = 0; k < KEIG; ++k) { sva[k] *= sign; rva[k] *= sign; }

    float* sp = sent_agg + (((size_t)b * NNODES + s) * KEIG) * DM + d;
    float* rp = recv_agg + (((size_t)b * NNODES + r) * KEIG) * DM + d;
#pragma unroll
    for (int k = 0; k < KEIG; ++k) {
        float eg = sva[k] * w0 + rva[k] * w1 + bb;
        eg = eg > 0.f ? eg : 0.f;
        atomicAdd(sp + k * DM, eg);
        atomicAdd(rp + k * DM, eg);
    }
}

// =====================================================================
// Kernel 2: node MLP + output linear, one wave per (b,n).
//   X[16,128] = [sent | recv];  H = relu(X@WnT + v*Wn0 + bn)   (16 WMMAs)
//   O = H@WoT + v*Wo0 + bo                                     (8 WMMAs)
// Weights live transposed in LDS -> B fragments are 2x ds_load_b128.
// =====================================================================
__global__ void __launch_bounds__(256)
node_mlp_kernel(const float* __restrict__ evec,   // [B,N,K]
                const float* __restrict__ sent,   // [B,N,K,64]
                const float* __restrict__ recv,   // [B,N,K,64]
                const float* __restrict__ Wn,     // [129,64]
                const float* __restrict__ bn,     // [64]
                const float* __restrict__ Wo,     // [65,64]
                const float* __restrict__ bo,     // [64]
                float* __restrict__ trans,        // [B,N,K,64]
                float sign, int accumulate) {
    __shared__ __align__(16) _Float16 sWnT[DM * 128];   // [col][krow] 16 KB
    __shared__ __align__(16) _Float16 sWoT[DM * 64];    // [col][krow]  8 KB
    __shared__ __align__(16) _Float16 sH[8][16 * DM];   // per-wave H   16 KB

    int tid = threadIdx.x;
    for (int i = tid; i < DM * 128; i += 256) {          // c = i>>7, rk = i&127
        int c = i >> 7, rk = i & 127;
        sWnT[i] = (_Float16)Wn[(1 + rk) * DM + c];
    }
    for (int i = tid; i < DM * 64; i += 256) {           // c = i>>6, rk = i&63
        int c = i >> 6, rk = i & 63;
        sWoT[i] = (_Float16)Wo[(1 + rk) * DM + c];
    }
    __syncthreads();

    int wave = tid >> 5, lane = tid & 31;
    int hi   = (lane >= 16) ? 1 : 0;
    int ln   = lane & 15;

    int p = blockIdx.x * 8 + wave;        // (b,n) pair in [0, B*N)

    const float* sbase = sent + (size_t)p * KEIG * DM;
    const float* rbase = recv + (size_t)p * KEIG * DM;

    // signed node scalar for the 8 rows this half-lane owns (2x b128)
    float evm[8];
    {
        const float4* e4 = (const float4*)(evec + (size_t)p * KEIG + hi * 8);
        float4 e0 = e4[0], e1 = e4[1];
        evm[0] = e0.x; evm[1] = e0.y; evm[2] = e0.z; evm[3] = e0.w;
        evm[4] = e1.x; evm[5] = e1.y; evm[6] = e1.z; evm[7] = e1.w;
        if (hi == 0) evm[0] = fabsf(evm[0]);            // global row 0
#pragma unroll
        for (int i = 0; i < 8; ++i) evm[i] *= sign;
    }

    // ---------------- GEMM1: [16,128] @ [128,64] ----------------
    v8f acc[4] = {v8f{}, v8f{}, v8f{}, v8f{}};
#pragma unroll
    for (int s = 0; s < 4; ++s) {                        // K steps of 32
        const float* src = (s < 2) ? sbase : rbase;      // cols 0..63 / 64..127
        int c0 = (s & 1) * 32 + hi * 8;
        const float4* q0 = (const float4*)(src + ln * DM + c0);
        const float4* q1 = (const float4*)(src + ln * DM + c0 + 16);
        v16h a = cvt16(q0[0], q0[1], q1[0], q1[1]);
#pragma unroll
        for (int t = 0; t < 4; ++t) {
            const v8h* bp = (const v8h*)&sWnT[(t * 16 + ln) * 128 + s * 32 + hi * 8];
            v16h bf = cat8(bp[0], bp[2]);
            acc[t] = wmma_f16(a, bf, acc[t]);
        }
    }

    // epilogue 1: + v*Wn0 + bn, relu, stage H (f16) in LDS
#pragma unroll
    for (int t = 0; t < 4; ++t) {
#pragma unroll
        for (int i = 0; i < 8; ++i) {
            int m = i + hi * 8;
            int c = t * 16 + ln;
            float hval = acc[t][i] + bn[c] + evm[i] * Wn[c];   // Wn row 0
            hval = hval > 0.f ? hval : 0.f;
            sH[wave][m * DM + c] = (_Float16)hval;
        }
    }
    __syncthreads();

    // ---------------- GEMM2: [16,64] @ [64,64] ----------------
    v8f acc2[4] = {v8f{}, v8f{}, v8f{}, v8f{}};
#pragma unroll
    for (int s = 0; s < 2; ++s) {
        const v8h* a0 = (const v8h*)&sH[wave][ln * DM + s * 32 + hi * 8];
        const v8h* a1 = (const v8h*)&sH[wave][ln * DM + s * 32 + 16 + hi * 8];
        v16h a = cat8(*a0, *a1);
#pragma unroll
        for (int t = 0; t < 4; ++t) {
            const v8h* bp = (const v8h*)&sWoT[(t * 16 + ln) * 64 + s * 32 + hi * 8];
            v16h bf = cat8(bp[0], bp[2]);
            acc2[t] = wmma_f16(a, bf, acc2[t]);
        }
    }

    // epilogue 2: + v*Wo0 + bo, write/accumulate trans
#pragma unroll
    for (int t = 0; t < 4; ++t) {
#pragma unroll
        for (int i = 0; i < 8; ++i) {
            int m = i + hi * 8;
            int c = t * 16 + ln;
            float val = acc2[t][i] + bo[c] + evm[i] * Wo[c];   // Wo row 0
            size_t oi = (size_t)p * KEIG * DM + m * DM + c;
            if (accumulate) trans[oi] += val;
            else            trans[oi]  = val;
        }
    }
}

// =====================================================================
// Kernel 3: masked re-aggregate GEMM  [16384,1024] @ [1024,256], bias+relu
// Register-blocked: one wave computes a 16x64 output strip (4 WMMA tiles),
// so each A fragment feeds 4 WMMAs and trans traffic drops 4x. 8 waves
// share a transposed 1024x64 Wa panel in dynamic LDS (128 KB; CDNA5 WGP
// allows up to 320 KB per workgroup). Eigenvalue mask is uniform per
// K-step -> skip whole K-steps (loads + 4 WMMAs) for masked components.
// =====================================================================
__global__ void __launch_bounds__(256)
final_gemm_kernel(const float* __restrict__ trans,   // [B*N, 1024]
                  const float* __restrict__ eigvals, // [B, K]
                  const float* __restrict__ Wa,      // [1024, 256]
                  const float* __restrict__ ba,      // [256]
                  float* __restrict__ out) {         // [B*N, 256]
    extern __shared__ __align__(16) _Float16 sBT[];  // [64 cols][1024 krows]

    int ngroup = blockIdx.x & 3;      // 4 groups of 64 output cols
    int mgroup = blockIdx.x >> 2;     // 128 groups of 128 rows
    int tid    = threadIdx.x;

    for (int i = tid; i < 64 * 1024; i += 256) {
        int c = i >> 10, rk = i & 1023;
        sBT[i] = (_Float16)Wa[(size_t)rk * DAGGR + ngroup * 64 + c];
    }
    __syncthreads();

    int wave = tid >> 5, lane = tid & 31;
    int hi = (lane >= 16) ? 1 : 0;
    int ln = lane & 15;

    int row0 = mgroup * 128 + wave * 16;     // 16 rows, same batch b
    int b    = row0 >> 12;

    unsigned mbits = 1u;                      // k=0 always active
#pragma unroll
    for (int kb = 1; kb < KEIG; ++kb)
        if (eigvals[b * KEIG + kb] > 0.f) mbits |= (1u << kb);

    const float* arow = trans + (size_t)(row0 + ln) * 1024;

    v8f acc[4] = {v8f{}, v8f{}, v8f{}, v8f{}};
    for (int s = 0; s < 32; ++s) {            // K = 1024, steps of 32
        if (!((mbits >> (s >> 1)) & 1u)) continue;   // masked eigencomponent
        int c0 = s * 32 + hi * 8;
        const float4* q0 = (const float4*)(arow + c0);
        const float4* q1 = (const float4*)(arow + c0 + 16);
        v16h a = cvt16(q0[0], q0[1], q1[0], q1[1]);
#pragma unroll
        for (int t = 0; t < 4; ++t) {
            const v8h* bp = (const v8h*)&sBT[(t * 16 + ln) * 1024 + s * 32 + hi * 8];
            v16h bf = cat8(bp[0], bp[2]);
            acc[t] = wmma_f16(a, bf, acc[t]);
        }
    }

#pragma unroll
    for (int t = 0; t < 4; ++t) {
#pragma unroll
        for (int i = 0; i < 8; ++i) {
            int m = i + hi * 8;
            int c = ngroup * 64 + t * 16 + ln;
            float v = acc[t][i] + ba[c];
            out[(size_t)(row0 + m) * DAGGR + c] = v > 0.f ? v : 0.f;
        }
    }
}

// =====================================================================
extern "C" void kernel_launch(void* const* d_in, const int* in_sizes, int n_in,
                              void* d_out, int out_size, void* d_ws, size_t ws_size,
                              hipStream_t stream) {
    const float* eigenvalues  = (const float*)d_in[0];   // [B,K]
    const float* eigenvectors = (const float*)d_in[1];   // [B,N,K]
    const int*   senders      = (const int*)d_in[2];     // [B,E]
    const int*   receivers    = (const int*)d_in[3];     // [B,E]
    const float* We = (const float*)d_in[4];
    const float* be = (const float*)d_in[5];
    const float* Wn = (const float*)d_in[6];
    const float* bn = (const float*)d_in[7];
    const float* Wo = (const float*)d_in[8];
    const float* bo = (const float*)d_in[9];
    const float* Wa = (const float*)d_in[10];
    const float* ba = (const float*)d_in[11];
    float* out = (float*)d_out;

    const size_t aggElems = (size_t)NB * NNODES * KEIG * DM;  // 16.78M
    float* sent  = (float*)d_ws;
    float* recvb = sent + aggElems;
    float* trans = recvb + aggElems;

    for (int pass = 0; pass < 2; ++pass) {
        float sign = pass ? -1.f : 1.f;
        hipMemsetAsync(sent, 0, 2 * aggElems * sizeof(float), stream);
        edge_kernel<<<NB * NE / 4, 256, 0, stream>>>(
            eigenvectors, senders, receivers, We, be, sent, recvb, sign);
        node_mlp_kernel<<<NB * NNODES / 8, 256, 0, stream>>>(
            eigenvectors, sent, recvb, Wn, bn, Wo, bo, trans, sign, pass);
    }
    // 4 col-groups x 128 row-groups; 128 KB dynamic LDS for the Wa panel
    final_gemm_kernel<<<4 * (NB * NNODES / 128), 256,
                        64 * 1024 * sizeof(_Float16), stream>>>(
        trans, eigenvalues, Wa, ba, out);
}
